// LearnableAttention_14937896255759
// MI455X (gfx1250) — compile-verified, matrix operand order checked by reading, not run on hardware
//
#include <hip/hip_runtime.h>
#include <hip/hip_bf16.h>
#include <math.h>

// ---------------------------------------------------------------------------
// LearnableAttention on MI455X (gfx1250): bf16 WMMA everywhere, fp32 accum.
// B=2, T=2048, C=1024, H=16, D=64
// Round 4: double-buffered async global->LDS staging (ASYNCcnt pipelining):
// issue tile i+1, s_wait_asynccnt <= inflight(i+1), compute tile i under the
// in-flight copies. GEMM K-step widened to 64.
// ---------------------------------------------------------------------------

#define BB 2
#define TT 2048
#define CC 1024
#define HH 16
#define DD 64
#define BT (BB * TT)   // 4096 rows

typedef __attribute__((ext_vector_type(16))) __bf16 v16bf;
typedef __attribute__((ext_vector_type(8)))  float  v8f;
typedef int b128_t __attribute__((vector_size(16)));   // int __vector(4)

union Frag16 { v16bf v; unsigned int u[8]; };

#if defined(__has_builtin)
#if __has_builtin(__builtin_amdgcn_global_load_async_to_lds_b128) && \
    __has_builtin(__builtin_amdgcn_s_wait_asynccnt)
#define HAVE_ASYNC_COPY 1
#endif
#endif

// 16-byte global -> LDS copy. Async (ASYNCcnt, no VGPR round trip) when the
// gfx1250 builtin exists; otherwise plain b128 load + ds_store.
__device__ __forceinline__ void copy16(const unsigned short* g, unsigned short* l) {
#ifdef HAVE_ASYNC_COPY
    __builtin_amdgcn_global_load_async_to_lds_b128(
        (__attribute__((address_space(1))) b128_t*)g,
        (__attribute__((address_space(3))) b128_t*)l, 0, 0);
#else
    *(uint4*)l = *(const uint4*)g;
#endif
}

__device__ __forceinline__ unsigned short f2bf(float f) {
    unsigned int x = __float_as_uint(f);
    return (unsigned short)((x + 0x7FFFu + ((x >> 16) & 1u)) >> 16);  // RNE
}

// K-pair start column for A/B bf16 fragments (16x32 / 32x16), per ISA 7.12.2:
// VGPR i in [0,3]: K = half*8 + i*2 ; VGPR i in [4,7]: K = 16 + half*8 + (i-4)*2
__device__ __forceinline__ int kkf(int i, int half) {
    return ((i & 4) << 2) + (half << 3) + ((i & 3) << 1);
}

__device__ __forceinline__ float hmax16(float v) {
    v = fmaxf(v, __shfl_xor(v, 1, 32));
    v = fmaxf(v, __shfl_xor(v, 2, 32));
    v = fmaxf(v, __shfl_xor(v, 4, 32));
    v = fmaxf(v, __shfl_xor(v, 8, 32));
    return v;
}
__device__ __forceinline__ float hsum16(float v) {
    v += __shfl_xor(v, 1, 32);
    v += __shfl_xor(v, 2, 32);
    v += __shfl_xor(v, 4, 32);
    v += __shfl_xor(v, 8, 32);
    return v;
}

// ---------------------------------------------------------------------------
// fp32 -> bf16 conversion (grid-stride)
// ---------------------------------------------------------------------------
__global__ void cvt_f32_bf16(const float* __restrict__ src,
                             unsigned short* __restrict__ dst, int n) {
    int i = blockIdx.x * blockDim.x + threadIdx.x;
    int stride = gridDim.x * blockDim.x;
    for (; i < n; i += stride) dst[i] = f2bf(src[i]);
}

// ---------------------------------------------------------------------------
// GEMM: 128 threads (4 waves) compute a 64x64 tile of y = x @ W^T.
// 64-wide K-steps; A/B 64x64 panels double-buffered in LDS via async copies.
// Per step: wave w owns rows [w*16,w*16+16), 2 A frags + 8 B frags, 8 WMMAs.
// ---------------------------------------------------------------------------
__device__ __forceinline__ void stage_gemm(const unsigned short* __restrict__ xb,
                                           const unsigned short* __restrict__ Wb,
                                           int t0, int o0, int c0,
                                           unsigned short* At, unsigned short* Bt) {
    const int tid = threadIdx.x;                // 0..127
#pragma unroll
    for (int p = 0; p < 4; ++p) {
        int idx = tid + p * 128;                // 0..511
        int row = idx >> 3;                     // 0..63
        int c8  = (idx & 7) * 8;                // 16B chunk within 64 cols
        copy16(xb + (size_t)(t0 + row) * CC + c0 + c8, At + row * 64 + c8);
        copy16(Wb + (size_t)(o0 + row) * CC + c0 + c8, Bt + row * 64 + c8);
    }
    // 8 async-copy instructions per wave per stage -> ASYNCcnt += 8
}

__device__ __forceinline__ void gemm_mainloop(const unsigned short* __restrict__ xb,
                                              const unsigned short* __restrict__ Wb,
                                              int t0, int o0,
                                              unsigned short (*At)[64 * 64],
                                              unsigned short (*Bt)[64 * 64],
                                              v8f acc[4]) {
    const int l = threadIdx.x & 31;
    const int w = threadIdx.x >> 5;
    const int n = l & 15;
    const int half = l >> 4;
    const int NSTEP = CC / 64;                  // 16

    stage_gemm(xb, Wb, t0, o0, 0, At[0], Bt[0]);

    for (int i = 0; i < NSTEP; ++i) {
        __syncthreads();                        // buf[(i+1)&1] readers (iter i-1) done
        if (i + 1 < NSTEP) {
            stage_gemm(xb, Wb, t0, o0, (i + 1) * 64, At[(i + 1) & 1], Bt[(i + 1) & 1]);
#ifdef HAVE_ASYNC_COPY
            __builtin_amdgcn_s_wait_asynccnt(8);   // tile i done; tile i+1 in flight
#endif
        } else {
#ifdef HAVE_ASYNC_COPY
            __builtin_amdgcn_s_wait_asynccnt(0);
#endif
        }
        __syncthreads();                        // all waves' tile-i copies landed

        const unsigned short* Ap = At[i & 1];
        const unsigned short* Bp = Bt[i & 1];
#pragma unroll
        for (int ds = 0; ds < 2; ++ds) {
            Frag16 A;
            const unsigned short* ar = Ap + (w * 16 + n) * 64 + ds * 32;
#pragma unroll
            for (int k = 0; k < 8; ++k)
                A.u[k] = *(const unsigned int*)(ar + kkf(k, half));
#pragma unroll
            for (int nb = 0; nb < 4; ++nb) {
                Frag16 Bf;
                const unsigned short* br = Bp + (nb * 16 + n) * 64 + ds * 32;
#pragma unroll
                for (int k = 0; k < 8; ++k)
                    Bf.u[k] = *(const unsigned int*)(br + kkf(k, half));
                acc[nb] = __builtin_amdgcn_wmma_f32_16x16x32_bf16(
                    false, A.v, false, Bf.v, (short)0, acc[nb], false, false);
            }
        }
    }
}

// QKV projection: y = x @ W^T + b, scattered into [B,H,T,D] bf16.
__global__ void gemm_qkv(const unsigned short* __restrict__ xb,
                         const unsigned short* __restrict__ Wb,
                         const float* __restrict__ bias,
                         unsigned short* __restrict__ dst) {
    __shared__ __align__(16) unsigned short At[2][64 * 64];
    __shared__ __align__(16) unsigned short Bt[2][64 * 64];
    const int t0 = blockIdx.x * 64;
    const int o0 = blockIdx.y * 64;
    const int w = threadIdx.x >> 5;
    const int l = threadIdx.x & 31;
    const int n = l & 15;
    const int half = l >> 4;

    v8f acc[4];
#pragma unroll
    for (int nb = 0; nb < 4; ++nb)
        acc[nb] = (v8f){0.f,0.f,0.f,0.f,0.f,0.f,0.f,0.f};

    gemm_mainloop(xb, Wb, t0, o0, At, Bt, acc);

#pragma unroll
    for (int nb = 0; nb < 4; ++nb)
#pragma unroll
        for (int j = 0; j < 8; ++j) {
            int row = t0 + w * 16 + j + half * 8;   // token in [0, BT)
            int o   = o0 + nb * 16 + n;             // channel
            float val = acc[nb][j] + bias[o];
            int b = row >> 11;
            int tt = row & (TT - 1);
            int h = o >> 6;
            int d = o & (DD - 1);
            dst[(((size_t)(b * HH + h)) * TT + tt) * DD + d] = f2bf(val);
        }
}

// Output projection: out = att @ Wp^T + bp, fp32 out.
__global__ void gemm_out(const unsigned short* __restrict__ ab,
                         const unsigned short* __restrict__ Wb,
                         const float* __restrict__ bias,
                         float* __restrict__ out) {
    __shared__ __align__(16) unsigned short At[2][64 * 64];
    __shared__ __align__(16) unsigned short Bt[2][64 * 64];
    const int t0 = blockIdx.x * 64;
    const int o0 = blockIdx.y * 64;
    const int w = threadIdx.x >> 5;
    const int l = threadIdx.x & 31;
    const int n = l & 15;
    const int half = l >> 4;

    v8f acc[4];
#pragma unroll
    for (int nb = 0; nb < 4; ++nb)
        acc[nb] = (v8f){0.f,0.f,0.f,0.f,0.f,0.f,0.f,0.f};

    gemm_mainloop(ab, Wb, t0, o0, At, Bt, acc);

#pragma unroll
    for (int nb = 0; nb < 4; ++nb)
#pragma unroll
        for (int j = 0; j < 8; ++j) {
            int row = t0 + w * 16 + j + half * 8;
            int o   = o0 + nb * 16 + n;
            out[(size_t)row * CC + o] = acc[nb][j] + bias[o];
        }
}

// ---------------------------------------------------------------------------
// Flash attention: 4 waves per block; wave w owns query tile (blockIdx.x*4+w),
// all waves share double-buffered 32-key K/V LDS tiles. S via 4 WMMAs, online
// softmax, P@V via 4 WMMAs per key tile.
// ---------------------------------------------------------------------------
__device__ __forceinline__ void stage_attn(const unsigned short* __restrict__ Kh,
                                           const unsigned short* __restrict__ Vh,
                                           int s0, unsigned short* Kt,
                                           unsigned short* Vt) {
    const int tid = threadIdx.x;                // 0..127
#pragma unroll
    for (int p = 0; p < 2; ++p) {
        int idx = tid + p * 128;                // 0..255
        int row = idx >> 3;                     // 0..31
        int c8  = (idx & 7) * 8;                // 16B chunk within 64 cols
        copy16(Kh + (size_t)(s0 + row) * DD + c8, Kt + row * DD + c8);
        copy16(Vh + (size_t)(s0 + row) * DD + c8, Vt + row * DD + c8);
    }
    // 4 async-copy instructions per wave per stage -> ASYNCcnt += 4
}

__global__ void attn_flash(const unsigned short* __restrict__ Q,
                           const unsigned short* __restrict__ K,
                           const unsigned short* __restrict__ V,
                           const float* __restrict__ mask,
                           unsigned short* __restrict__ att) {
    const int bh = blockIdx.y;                 // b*H + h
    const int b = bh >> 4;
    const int h = bh & 15;

    const size_t base = (size_t)bh * TT * DD;
    const unsigned short* Qh = Q + base;
    const unsigned short* Kh = K + base;
    const unsigned short* Vh = V + base;

    const int tid = threadIdx.x;               // 0..127
    const int w = tid >> 5;
    const int l = tid & 31;
    const int n = l & 15;
    const int half = l >> 4;
    const int q0 = blockIdx.x * 64 + w * 16;   // this wave's query rows

    __shared__ __align__(16) unsigned short Kt[2][32 * DD];
    __shared__ __align__(16) unsigned short Vt[2][32 * DD];
    __shared__ __align__(16) unsigned short Pt[4][16 * 32];

    // Q A-fragments for d=0..31 and d=32..63
    Frag16 aq[2];
    {
        const unsigned short* qrow = Qh + (size_t)(q0 + n) * DD;
#pragma unroll
        for (int ds = 0; ds < 2; ++ds)
#pragma unroll
            for (int i = 0; i < 8; ++i)
                aq[ds].u[i] = *(const unsigned int*)(qrow + ds * 32 + kkf(i, half));
    }

    const float scale = 0.125f;                // 1/sqrt(64)
    const float maskv = mask[0];

    float mi[8], li[8];
    v8f accd[4];
#pragma unroll
    for (int j = 0; j < 8; ++j) { mi[j] = -INFINITY; li[j] = 0.f; }
#pragma unroll
    for (int d = 0; d < 4; ++d)
        accd[d] = (v8f){0.f,0.f,0.f,0.f,0.f,0.f,0.f,0.f};

    const int NS = TT / 32;                    // 64 key tiles
    stage_attn(Kh, Vh, 0, Kt[0], Vt[0]);

    for (int it = 0; it < NS; ++it) {
        __syncthreads();                       // buf[(it+1)&1] readers done
        if (it + 1 < NS) {
            stage_attn(Kh, Vh, (it + 1) * 32, Kt[(it + 1) & 1], Vt[(it + 1) & 1]);
#ifdef HAVE_ASYNC_COPY
            __builtin_amdgcn_s_wait_asynccnt(4);  // tile it done; it+1 in flight
#endif
        } else {
#ifdef HAVE_ASYNC_COPY
            __builtin_amdgcn_s_wait_asynccnt(0);
#endif
        }
        __syncthreads();                       // all waves' tile-it copies landed

        const unsigned short* Kp = Kt[it & 1];
        const unsigned short* Vp = Vt[it & 1];

        // ---- S = Q @ K^T (16 x 32), two 16-key column blocks ----
        v8f s[2];
#pragma unroll
        for (int kb = 0; kb < 2; ++kb) {
            s[kb] = (v8f){0.f,0.f,0.f,0.f,0.f,0.f,0.f,0.f};
#pragma unroll
            for (int ds = 0; ds < 2; ++ds) {
                Frag16 bk;                     // B[d][k] = K[s0+kb*16+k][d]
                const unsigned short* krow = Kp + (kb * 16 + n) * DD + ds * 32;
#pragma unroll
                for (int i = 0; i < 8; ++i)
                    bk.u[i] = *(const unsigned int*)(krow + kkf(i, half));
                s[kb] = __builtin_amdgcn_wmma_f32_16x16x32_bf16(
                    false, aq[ds].v, false, bk.v, (short)0, s[kb], false, false);
            }
#pragma unroll
            for (int j = 0; j < 8; ++j)
                s[kb][j] = s[kb][j] * scale + maskv;
        }

        // ---- online softmax (rows live across 16-lane groups) ----
        float al[8];
#pragma unroll
        for (int j = 0; j < 8; ++j) {
            float r = hmax16(fmaxf(s[0][j], s[1][j]));
            float mnew = fmaxf(mi[j], r);
            s[0][j] = __expf(s[0][j] - mnew);
            s[1][j] = __expf(s[1][j] - mnew);
            float rs = hsum16(s[0][j] + s[1][j]);
            float a = __expf(mi[j] - mnew);
            li[j] = li[j] * a + rs;
            mi[j] = mnew;
            al[j] = a;
        }
#pragma unroll
        for (int d = 0; d < 4; ++d)
#pragma unroll
            for (int j = 0; j < 8; ++j) accd[d][j] *= al[j];

        // ---- re-layout P: C/D frag -> per-wave LDS buffer -> A frag ----
#pragma unroll
        for (int kb = 0; kb < 2; ++kb)
#pragma unroll
            for (int j = 0; j < 8; ++j)
                Pt[w][(j + half * 8) * 32 + kb * 16 + n] = f2bf(s[kb][j]);
        __syncthreads();

        Frag16 pa;
        {
            const unsigned short* prow = Pt[w] + n * 32;
#pragma unroll
            for (int i = 0; i < 8; ++i)
                pa.u[i] = *(const unsigned int*)(prow + kkf(i, half));
        }

        // ---- O += P @ V, four 16-wide d blocks ----
#pragma unroll
        for (int d = 0; d < 4; ++d) {
            Frag16 bv;                         // B[k][n] = V[s0+k][d*16+n]
#pragma unroll
            for (int i = 0; i < 8; ++i) {
                int kk = kkf(i, half);
                unsigned int lo = Vp[kk * DD + d * 16 + n];
                unsigned int hi = Vp[(kk + 1) * DD + d * 16 + n];
                bv.u[i] = lo | (hi << 16);
            }
            accd[d] = __builtin_amdgcn_wmma_f32_16x16x32_bf16(
                false, pa.v, false, bv.v, (short)0, accd[d], false, false);
        }
    }

    // epilogue: normalize, scatter to [B,T,C] bf16
#pragma unroll
    for (int d = 0; d < 4; ++d)
#pragma unroll
        for (int j = 0; j < 8; ++j) {
            int row = q0 + j + half * 8;
            int c = h * DD + d * 16 + n;
            float o = accd[d][j] / li[j];
            att[((size_t)(b * TT + row)) * CC + c] = f2bf(o);
        }
}

// ---------------------------------------------------------------------------
// Launch
// ---------------------------------------------------------------------------
extern "C" void kernel_launch(void* const* d_in, const int* in_sizes, int n_in,
                              void* d_out, int out_size, void* d_ws, size_t ws_size,
                              hipStream_t stream) {
    const float* x    = (const float*)d_in[0];
    const float* Wq   = (const float*)d_in[1];
    const float* bq   = (const float*)d_in[2];
    const float* Wk   = (const float*)d_in[3];
    const float* bk   = (const float*)d_in[4];
    const float* Wv   = (const float*)d_in[5];
    const float* bv   = (const float*)d_in[6];
    const float* mask = (const float*)d_in[7];
    const float* Wp   = (const float*)d_in[8];
    const float* bp   = (const float*)d_in[9];
    float* out = (float*)d_out;

    char* ws = (char*)d_ws;
    const size_t MB = 1024 * 1024;
    unsigned short* xb  = (unsigned short*)(ws + 0);        // 8 MB  [BT, C]
    unsigned short* wqb = (unsigned short*)(ws + 8  * MB);  // 2 MB
    unsigned short* wkb = (unsigned short*)(ws + 10 * MB);  // 2 MB
    unsigned short* wvb = (unsigned short*)(ws + 12 * MB);  // 2 MB
    unsigned short* wpb = (unsigned short*)(ws + 14 * MB);  // 2 MB
    unsigned short* Qb  = (unsigned short*)(ws + 16 * MB);  // 8 MB  [B,H,T,D]
    unsigned short* Kb  = (unsigned short*)(ws + 24 * MB);  // 8 MB
    unsigned short* Vb  = (unsigned short*)(ws + 32 * MB);  // 8 MB
    unsigned short* Ab  = (unsigned short*)(ws + 40 * MB);  // 8 MB  [BT, C]

    // 1) fp32 -> bf16
    cvt_f32_bf16<<<2048, 256, 0, stream>>>(x,  xb,  BT * CC);
    cvt_f32_bf16<<<1024, 256, 0, stream>>>(Wq, wqb, CC * CC);
    cvt_f32_bf16<<<1024, 256, 0, stream>>>(Wk, wkb, CC * CC);
    cvt_f32_bf16<<<1024, 256, 0, stream>>>(Wv, wvb, CC * CC);
    cvt_f32_bf16<<<1024, 256, 0, stream>>>(Wp, wpb, CC * CC);

    // 2) QKV projections (WMMA, 64x64 tiles, double-buffered async staging)
    dim3 ggrid(BT / 64, CC / 64);
    gemm_qkv<<<ggrid, 128, 0, stream>>>(xb, wqb, bq, Qb);
    gemm_qkv<<<ggrid, 128, 0, stream>>>(xb, wkb, bk, Kb);
    gemm_qkv<<<ggrid, 128, 0, stream>>>(xb, wvb, bv, Vb);

    // 3) flash attention (WMMA), 64 queries/block, -> [B,T,C] bf16
    attn_flash<<<dim3(TT / 64, BB * HH), 128, 0, stream>>>(Qb, Kb, Vb, mask, Ab);

    // 4) output projection (WMMA), fp32 out
    gemm_out<<<ggrid, 128, 0, stream>>>(Ab, wpb, bp, out);
}
